// MADpredictor_21199958573258
// MI455X (gfx1250) — matrix operand order repaired and checked by reading, not compiled, not run on hardware
//
#include <hip/hip_runtime.h>
#include <hip/hip_bf16.h>
#include <stdint.h>

#define H_ 4
#define N_ 10000
#define D_ 128
#define B_ 2048
#define K_ 8
#define NT_ 625   // N_/16 column tiles
#define BT_ 64    // B_/32 batch tiles (32 rows per workgroup)

typedef __attribute__((ext_vector_type(16))) __bf16 v16bf;
typedef __attribute__((ext_vector_type(8)))  float  v8f;

union BfVec { uint32_t u[8]; v16bf v; };

// ---------------------------------------------------------------------------
// Kernel 1: f32 -> bf16 (RNE) conversion + per-row squared norms
// grid = H_*N_ blocks of 128 threads (one row of D=128 each)
// ---------------------------------------------------------------------------
__global__ __launch_bounds__(128) void mad_prep_kernel(
    const float* __restrict__ embeds, unsigned short* __restrict__ ebf,
    float* __restrict__ n2) {
  const int row = blockIdx.x;          // h*N_ + n
  const int t   = threadIdx.x;         // 0..127
  float v = embeds[(size_t)row * D_ + t];
  uint32_t u = __float_as_uint(v);
  uint32_t r = (u + 0x7FFFu + ((u >> 16) & 1u)) >> 16;   // round-to-nearest-even
  ebf[(size_t)row * D_ + t] = (unsigned short)r;
  float s = v * v;
  #pragma unroll
  for (int o = 16; o > 0; o >>= 1) s += __shfl_xor(s, o, 32);
  __shared__ float ws4[4];
  if ((t & 31) == 0) ws4[t >> 5] = s;
  __syncthreads();
  if (t == 0) n2[row] = ws4[0] + ws4[1] + ws4[2] + ws4[3];
}

// ---------------------------------------------------------------------------
// Kernel 2: fused WMMA distance GEMM (M=32 per WG) + ballot-gated top-(K+1).
// grid = 2*H_*BT_ blocks (dir,h,32-row batch tile) of 128 threads (4 wave32).
// ---------------------------------------------------------------------------
__global__ __launch_bounds__(128) void mad_knn_kernel(
    const unsigned short* __restrict__ ebf, const float* __restrict__ n2,
    const long long* __restrict__ edges, int* __restrict__ knn_idx) {
  int blk  = blockIdx.x;
  const int tile = blk & (BT_ - 1);
  const int h    = (blk >> 6) & 3;
  const int dir  = blk >> 8;
  const int t    = threadIdx.x;
  const int wave = t >> 5;
  const int lane = t & 31;
  const int laneLo = lane & 15;
  const bool hi  = lane >= 16;
  const int M0   = hi ? 8 : 0;

  __shared__ float s_sc[4][32][18];    // padded rows: conflict-free
  __shared__ float s_md[4][32][9];
  __shared__ int   s_mi[4][32][9];
  __shared__ int   s_node[32];
  __shared__ float s_q2[32];

  if (t < 32) {
    int b = tile * 32 + t;
    int node = (int)edges[(size_t)dir * B_ + b];
    s_node[t] = node;
    s_q2[t]   = n2[h * N_ + node];
  }
  __syncthreads();

  // ---- Load two A (Q) tiles once, in contiguous k order k = 16*(lane/16) + s, 4 k-steps of 32 --
  v16bf a0[4], a1[4];
  #pragma unroll
  for (int half = 0; half < 2; ++half) {
    const unsigned short* rp =
        ebf + ((size_t)(h * N_ + s_node[half * 16 + laneLo])) * D_;
    #pragma unroll
    for (int ks = 0; ks < 4; ++ks) {
      BfVec bv;
      const uint32_t* p0 = (const uint32_t*)(rp + ks * 32 + (hi ? 16 : 0));
      const uint32_t* p1 = (const uint32_t*)(rp + ks * 32 + 8 + (hi ? 16 : 0));
      #pragma unroll
      for (int q = 0; q < 4; ++q) { bv.u[q] = p0[q]; bv.u[4 + q] = p1[q]; }
      if (half == 0) a0[ks] = bv.v; else a1[ks] = bv.v;
    }
  }
  const float q2v = s_q2[lane];        // every lane owns row `lane`

  // per-lane sorted top-9 (ascending d2) for its own row
  float bd[9]; int bi[9];
  #pragma unroll
  for (int k = 0; k < 9; ++k) { bd[k] = 3.0e38f; bi[k] = 0; }

  // wave-uniform conservative slack threshold on s = n2 - 2*dot
  // (recomputed only when an insertion happens; init forces first trigger)
  float thr = 3.0e38f;

  const float* n2h = n2 + h * N_;
  const unsigned short* ebh = ebf + (size_t)h * N_ * D_;

  for (int j = wave; j < NT_; j += 4) {
    const int n0 = j * 16;
    if (j + 4 < NT_)
      __builtin_prefetch(ebh + ((size_t)(n0 + 64 + laneLo)) * D_, 0, 1);

    // ---- Load B (16 database columns), contiguous k order k = 16*(lane/16) + s, same as A ----
    v16bf bmat[4];
    const unsigned short* cp = ebh + ((size_t)(n0 + laneLo)) * D_;
    #pragma unroll
    for (int ks = 0; ks < 4; ++ks) {
      BfVec bv;
      const uint32_t* p = (const uint32_t*)(cp + ks * 32 + (hi ? 16 : 0));
      #pragma unroll
      for (int q = 0; q < 8; ++q) bv.u[q] = p[q];
      bmat[ks] = bv.v;
    }

    // ---- 32x16 dot tile via 8x v_wmma_f32_16x16x32_bf16 (A reuse) ----
    v8f c0 = {}, c1 = {};
    #pragma unroll
    for (int ks = 0; ks < 4; ++ks) {
      c0 = __builtin_amdgcn_wmma_f32_16x16x32_bf16(
               false, a0[ks], false, bmat[ks], (short)0, c0, false, false);
      c1 = __builtin_amdgcn_wmma_f32_16x16x32_bf16(
               false, a1[ks], false, bmat[ks], (short)0, c1, false, false);
    }

    // fold ||n||^2 - 2*dot (lane owns column N = laneLo of the score tile)
    const float n2v = n2h[n0 + laneLo];
    float sc0[8], sc1[8];
    #pragma unroll
    for (int r = 0; r < 8; ++r) {
      sc0[r] = fmaf(-2.0f, c0[r], n2v);
      sc1[r] = fmaf(-2.0f, c1[r], n2v);
    }

    // ---- register-only candidate test against cached wave threshold ----
    float mn = sc0[0];
    #pragma unroll
    for (int r = 1; r < 8; ++r) mn = fminf(mn, sc0[r]);
    #pragma unroll
    for (int r = 0; r < 8; ++r) mn = fminf(mn, sc1[r]);

    if (__builtin_amdgcn_ballot_w32(mn < thr)) {   // scalar branch: rare path
      #pragma unroll
      for (int r = 0; r < 8; ++r) {
        s_sc[wave][M0 + r][laneLo]      = sc0[r];
        s_sc[wave][16 + M0 + r][laneLo] = sc1[r];
      }
      asm volatile("s_wait_dscnt 0" ::: "memory");
      // every lane scans its own row with exact insertion
      #pragma unroll 4
      for (int cc = 0; cc < 16; ++cc) {
        float d2 = q2v + s_sc[wave][lane][cc];
        if (d2 < bd[8]) {
          bd[8] = d2; bi[8] = n0 + cc;
          #pragma unroll
          for (int s = 8; s > 0; --s) {
            if (bd[s] < bd[s - 1]) {
              float tf = bd[s]; bd[s] = bd[s - 1]; bd[s - 1] = tf;
              int   ti = bi[s]; bi[s] = bi[s - 1]; bi[s - 1] = ti;
            }
          }
        }
      }
      // refresh conservative wave-max threshold (uniform path)
      float t8 = bd[8] - q2v;
      #pragma unroll
      for (int o = 16; o > 0; o >>= 1) t8 = fmaxf(t8, __shfl_xor(t8, o, 32));
      thr = t8;
      asm volatile("" ::: "memory");
    }
  }

  // ---- merge the 4 per-wave partial lists ----
  #pragma unroll
  for (int k = 0; k < 9; ++k) { s_md[wave][lane][k] = bd[k]; s_mi[wave][lane][k] = bi[k]; }
  __syncthreads();

  if (t < 32) {
    float md[9]; int mi[9];
    #pragma unroll
    for (int k = 0; k < 9; ++k) { md[k] = s_md[0][t][k]; mi[k] = s_mi[0][t][k]; }
    for (int w = 1; w < 4; ++w) {
      #pragma unroll
      for (int k = 0; k < 9; ++k) {
        float d = s_md[w][t][k]; int i = s_mi[w][t][k];
        if (d < md[8]) {
          md[8] = d; mi[8] = i;
          #pragma unroll
          for (int s = 8; s > 0; --s) {
            if (md[s] < md[s - 1]) {
              float tf = md[s]; md[s] = md[s - 1]; md[s - 1] = tf;
              int   ti = mi[s]; mi[s] = mi[s - 1]; mi[s - 1] = ti;
            }
          }
        }
      }
    }
    const int b = tile * 32 + t;
    int* op = knn_idx + ((size_t)((dir * H_ + h) * B_ + b)) * K_;
    #pragma unroll
    for (int k = 0; k < K_; ++k) op[k] = mi[k + 1];   // drop self (rank 0)
  }
}

// ---------------------------------------------------------------------------
// Kernel 3: per-(dir,h,b): logits[k] = diff.field + u*adj[...], dist[k]=|diff|
// grid = 2*H_*B_ blocks of 128 threads (thread t: k = t/16, 8 elems of D each)
// ---------------------------------------------------------------------------
__global__ __launch_bounds__(128) void mad_edge_kernel(
    const float* __restrict__ embeds, const float* __restrict__ field,
    const float* __restrict__ adj, const float* __restrict__ unc,
    const long long* __restrict__ edges, const int* __restrict__ knn_idx,
    float* __restrict__ logits, float* __restrict__ dist) {
  int blk = blockIdx.x;
  const int b   = blk & (B_ - 1);
  const int h   = (blk >> 11) & 3;
  const int dir = blk >> 13;
  const int t   = threadIdx.x;
  const int k   = t >> 4;
  const int sub = t & 15;

  const int node = (int)edges[(size_t)dir * B_ + b];
  const int nb   = knn_idx[((size_t)((dir * H_ + h) * B_ + b)) * K_ + k];
  const float* qrow = embeds + ((size_t)(h * N_ + node)) * D_;
  const float* nrow = embeds + ((size_t)(h * N_ + nb)) * D_;
  const float* frow = field  + ((size_t)(h * N_ + node)) * D_;

  float sdot = 0.0f, sn = 0.0f;
  #pragma unroll
  for (int d = sub; d < D_; d += 16) {
    float df = qrow[d] - nrow[d];
    sdot = fmaf(df, frow[d], sdot);
    sn   = fmaf(df, df, sn);
  }
  #pragma unroll
  for (int o = 8; o > 0; o >>= 1) {
    sdot += __shfl_xor(sdot, o, 16);
    sn   += __shfl_xor(sn,   o, 16);
  }
  if (sub == 0) {
    float label = unc[0] * (dir == 0 ? adj[(size_t)nb * N_ + node]
                                     : adj[(size_t)node * N_ + nb]);
    size_t o = ((size_t)((dir * H_ + h) * B_ + b)) * K_ + k;
    logits[o] = sdot + label;
    dist[o]   = sqrtf(sn);
  }
}

// ---------------------------------------------------------------------------
// Kernel 4: per-b softmax over 24 entries (16 real + 8 pads), head mean, sigmoid
// ---------------------------------------------------------------------------
__global__ __launch_bounds__(256) void mad_reduce_kernel(
    const float* __restrict__ logits, const float* __restrict__ dist,
    float* __restrict__ out) {
  int b = blockIdx.x * blockDim.x + threadIdx.x;
  if (b >= B_) return;
  float acc = 0.0f;
  #pragma unroll
  for (int h = 0; h < H_; ++h) {
    float l[16], dd[16];
    #pragma unroll
    for (int dir = 0; dir < 2; ++dir)
      #pragma unroll
      for (int k = 0; k < K_; ++k) {
        size_t o = ((size_t)((dir * H_ + h) * B_ + b)) * K_ + k;
        l[dir * K_ + k]  = logits[o];
        dd[dir * K_ + k] = dist[o];
      }
    float m = 0.0f;                       // pads give score 1-1 = 0
    #pragma unroll
    for (int i = 0; i < 16; ++i) m = fmaxf(m, 1.0f - dd[i]);
    float se = 8.0f * __expf(-m);         // 8 pads, logit 0
    float sl = 0.0f;
    #pragma unroll
    for (int i = 0; i < 16; ++i) {
      float e = __expf(1.0f - dd[i] - m);
      se += e;
      sl = fmaf(l[i], e, sl);
    }
    acc += sl / se;
  }
  acc *= (1.0f / H_);
  out[b] = 1.0f / (1.0f + __expf(-acc));
}

// ---------------------------------------------------------------------------
extern "C" void kernel_launch(void* const* d_in, const int* in_sizes, int n_in,
                              void* d_out, int out_size, void* d_ws, size_t ws_size,
                              hipStream_t stream) {
  (void)in_sizes; (void)n_in; (void)out_size; (void)ws_size;
  const float*     embeds = (const float*)d_in[0];
  const long long* edges  = (const long long*)d_in[1];
  const float*     adj    = (const float*)d_in[2];
  const float*     field  = (const float*)d_in[3];
  const float*     unc    = (const float*)d_in[4];
  float* out = (float*)d_out;

  char* ws = (char*)d_ws;
  unsigned short* ebf = (unsigned short*)ws;                    // 10,240,000 B
  float* n2   = (float*)(ws + 10240000);                        //    160,000 B
  int*   knn  = (int*)  (ws + 10400000);                        //    524,288 B
  float* lg   = (float*)(ws + 10924288);                        //    524,288 B
  float* dst  = (float*)(ws + 11448576);                        //    524,288 B

  mad_prep_kernel<<<H_ * N_, 128, 0, stream>>>(embeds, ebf, n2);
  mad_knn_kernel<<<2 * H_ * BT_, 128, 0, stream>>>(ebf, n2, edges, knn);
  mad_edge_kernel<<<2 * H_ * B_, 128, 0, stream>>>(embeds, field, adj, unc,
                                                   edges, knn, lg, dst);
  mad_reduce_kernel<<<(B_ + 255) / 256, 256, 0, stream>>>(lg, dst, out);
}
